// Block_47820165874437
// MI455X (gfx1250) — matrix operand order repaired and checked by reading, not compiled, not run on hardware
//
#include <hip/hip_runtime.h>

// ---------------- problem constants ----------------
#define NPIX   16384      // H*W = 128*128
#define BATCH  8
#define CCH    128
#define NHEADS 8
#define MIDCH  340
#define O_QKV  384
#define O_PW12 688        // 680 padded up to multiple of 16
#define K_PWO  352        // 340 padded up to multiple of 32

typedef __bf16 bf16_t;
typedef bf16_t v16bf __attribute__((ext_vector_type(16)));
typedef float  v8f   __attribute__((ext_vector_type(8)));

union V16 { v16bf v; uint4 u4[2]; unsigned short us[16]; };

#if defined(__has_builtin)
#if __has_builtin(__builtin_amdgcn_global_load_async_to_lds_b32) && \
    __has_builtin(__builtin_amdgcn_s_wait_asynccnt)
#define USE_ASYNC_LDS 1
#endif
#endif

#define AS_GLOBAL __attribute__((address_space(1)))
#define AS_LDS    __attribute__((address_space(3)))

__device__ __forceinline__ unsigned short f2bf(float f) {
  unsigned u = __float_as_uint(f);
  return (unsigned short)((u + 0x7FFFu + ((u >> 16) & 1u)) >> 16);  // RNE
}
__device__ __forceinline__ float bf2f(unsigned short h) {
  return __uint_as_float(((unsigned)h) << 16);
}

// ---------------- weight convert / pad (fp32 -> bf16) ----------------
__global__ void cvt_pad(const float* __restrict__ src, unsigned short* __restrict__ dst,
                        int dr, int dc, int sr, int sc) {
  int i = blockIdx.x * 256 + threadIdx.x;
  if (i >= dr * dc) return;
  int r = i / dc, c = i - r * dc;
  float v = (r < sr && c < sc) ? src[r * sc + c] : 0.f;
  dst[i] = f2bf(v);
}

// ---------------- GroupNorm(1,C): stats pass ----------------
__global__ void gn_stats(const float* __restrict__ x, float* __restrict__ stats) {
  __shared__ float s1[256], s2[256];
  int b = blockIdx.y;
  size_t base = (size_t)b * (CCH * (size_t)NPIX) + (size_t)blockIdx.x * 4096 + threadIdx.x * 16;
  const float4* p = (const float4*)(x + base);
  float sum = 0.f, sq = 0.f;
#pragma unroll
  for (int i = 0; i < 4; ++i) {
    float4 v = p[i];
    sum += v.x + v.y + v.z + v.w;
    sq  += v.x * v.x + v.y * v.y + v.z * v.z + v.w * v.w;
  }
  s1[threadIdx.x] = sum; s2[threadIdx.x] = sq;
  __syncthreads();
  for (int st = 128; st > 0; st >>= 1) {
    if (threadIdx.x < st) { s1[threadIdx.x] += s1[threadIdx.x + st]; s2[threadIdx.x] += s2[threadIdx.x + st]; }
    __syncthreads();
  }
  if (threadIdx.x == 0) {
    atomicAdd(&stats[b * 2 + 0], s1[0]);
    atomicAdd(&stats[b * 2 + 1], s2[0]);
  }
}

// ---------------- GroupNorm apply + cast to bf16 ----------------
__global__ void gn_apply(const float* __restrict__ x, const float* __restrict__ stats,
                         const float* __restrict__ w, const float* __restrict__ bwt,
                         unsigned short* __restrict__ dst) {
  size_t gid = (size_t)blockIdx.x * 256 + threadIdx.x;
  size_t e0 = gid * 4;
  int b = (int)(e0 >> 21);            // C*N = 2^21
  int c = (int)((e0 >> 14) & 127);    // N   = 2^14
  const float inv = 1.f / 2097152.f;
  float mean = stats[b * 2 + 0] * inv;
  float var  = stats[b * 2 + 1] * inv - mean * mean;
  float rstd = rsqrtf(var + 1e-5f);
  float sc = w[c] * rstd;
  float sh = bwt[c] - mean * sc;
  float4 v = *(const float4*)(x + e0);
  ushort4 o;
  o.x = f2bf(v.x * sc + sh); o.y = f2bf(v.y * sc + sh);
  o.z = f2bf(v.z * sc + sh); o.w = f2bf(v.w * sc + sh);
  *(ushort4*)(dst + e0) = o;
}

// ---------------- WMMA GEMM: OUT[b][o][n] = sum_k W[o][k]*ACT[b][k][n] ----------------
// block = 256 thr (8 waves); block tile = 128(O) x 64(N); K multiple of 32.
// mode 0: store bf16 to out_bf.  mode 1: out_f = res + scale * acc (fp32).
__global__ void gemm_wmma(const unsigned short* __restrict__ W,
                          const unsigned short* __restrict__ act,
                          int K, int Opad,
                          unsigned short* __restrict__ out_bf,
                          float* __restrict__ out_f,
                          const float* __restrict__ res,
                          const float* __restrict__ scale_p,
                          int mode) {
  __shared__ unsigned short ldsB[64 * 40];   // [n][k], stride 40 halves -> 16B-aligned lane reads
  int tid  = threadIdx.x;
  int wave = tid >> 5, lane = tid & 31;
  int sel  = lane >> 4, lr = lane & 15;
  int b  = blockIdx.z;
  int n0 = blockIdx.x * 64;
  int o0 = blockIdx.y * 128;
  const unsigned short* actb = act + (size_t)b * K * (size_t)NPIX;
  int m = o0 + wave * 16 + lr;
  int mclamp = (m < Opad) ? m : (Opad - 1);
  const unsigned short* wrow = W + (size_t)mclamp * K;

  v8f z;
#pragma unroll
  for (int i = 0; i < 8; ++i) z[i] = 0.f;
  v8f acc[4] = { z, z, z, z };

  int kk2 = (tid >> 4) * 2;     // even k row: 0..30
  int n4  = (tid & 15) * 4;     // n within tile: 0..60
  unsigned* lp = (unsigned*)&ldsB[0];

  for (int kb = 0; kb < K; kb += 32) {
    // stage ACT tile [32k x 64n], transposed, k-pairs packed into b32 LDS stores
    uint2 r0 = *(const uint2*)(actb + (size_t)(kb + kk2)     * NPIX + n0 + n4);
    uint2 r1 = *(const uint2*)(actb + (size_t)(kb + kk2 + 1) * NPIX + n0 + n4);
    if (kb + 32 < K)            // prefetch next K-step tile (global_prefetch_b8)
      __builtin_prefetch(actb + (size_t)(kb + 32 + kk2) * NPIX + n0 + n4, 0, 0);
    __syncthreads();            // previous tile fully consumed
    const unsigned short* e0 = (const unsigned short*)&r0;
    const unsigned short* e1 = (const unsigned short*)&r1;
#pragma unroll
    for (int j = 0; j < 4; ++j) {
      unsigned packed = (unsigned)e0[j] | ((unsigned)e1[j] << 16);  // (k,k+1) @ column n4+j
      lp[(n4 + j) * 20 + (kk2 >> 1)] = packed;
    }
    __syncthreads();

    V16 a;                      // A: 16x32 bf16 per ISA lane layout
    a.u4[0] = *(const uint4*)(wrow + kb + sel * 8);
    a.u4[1] = *(const uint4*)(wrow + kb + sel * 8 + 16);
    V16 bm[4];                  // load all 4 B tiles first -> incremental dscnt waits
#pragma unroll
    for (int t = 0; t < 4; ++t) {
      const unsigned short* pb = &ldsB[(t * 16 + lr) * 40 + sel * 8];
      bm[t].u4[0] = *(const uint4*)pb;
      bm[t].u4[1] = *(const uint4*)(pb + 16);
    }
#pragma unroll
    for (int t = 0; t < 4; ++t)
      acc[t] = __builtin_amdgcn_wmma_f32_16x16x32_bf16(false, a.v, false, bm[t].v,
                                                       (short)0, acc[t], false, false);
  }

  float scale = mode ? scale_p[0] : 0.f;
#pragma unroll
  for (int t = 0; t < 4; ++t) {
#pragma unroll
    for (int r = 0; r < 8; ++r) {
      int mm = o0 + wave * 16 + r + sel * 8;        // C/D layout: VGPR r -> M=r (+8 for hi lanes)
      if (mm >= Opad) continue;
      int nc = n0 + t * 16 + lr;
      size_t idx = ((size_t)b * Opad + mm) * NPIX + nc;
      float v = acc[t][r];
      if (mode) out_f[idx] = res[idx] + scale * v;
      else      out_bf[idx] = f2bf(v);
    }
  }
}

// ---------------- depthwise 3x3 on qkv + q/k row sum-of-squares ----------------
__global__ void dw3_qkv(const unsigned short* __restrict__ src,
                        const float* __restrict__ wdw,
                        unsigned short* __restrict__ dst,
                        float* __restrict__ qknorm2) {
  __shared__ float red[256];
  int tid = threadIdx.x;
  int c = blockIdx.y, b = blockIdx.z;
  int n = blockIdx.x * 256 + tid;
  int y = n >> 7, xx = n & 127;
  const unsigned short* row = src + ((size_t)b * O_QKV + c) * NPIX;
  const float* wc = wdw + c * 9;
  float acc = 0.f;
#pragma unroll
  for (int dy = -1; dy <= 1; ++dy) {
    int yy = y + dy;
    if (yy < 0 || yy > 127) continue;
#pragma unroll
    for (int dx = -1; dx <= 1; ++dx) {
      int xb = xx + dx;
      if (xb < 0 || xb > 127) continue;
      acc += wc[(dy + 1) * 3 + (dx + 1)] * bf2f(row[yy * 128 + xb]);
    }
  }
  dst[((size_t)b * O_QKV + c) * NPIX + n] = f2bf(acc);
  if (c < 256) {                                   // q and k channels only
    red[tid] = acc * acc;
    __syncthreads();
    for (int st = 128; st > 0; st >>= 1) {
      if (tid < st) red[tid] += red[tid + st];
      __syncthreads();
    }
    if (tid == 0) atomicAdd(&qknorm2[b * 256 + c], red[0]);
  }
}

// ---------------- attention: S = q.k^T via WMMA (K=16384), l2norm folded, softmax -> P ----------------
__global__ void attn_qk(const unsigned short* __restrict__ qkv,
                        const float* __restrict__ qknorm2,
                        const float* __restrict__ temp_p,
                        float* __restrict__ P) {
  __shared__ float spart[8 * 256];
  __shared__ float sfin[256];
  int tid = threadIdx.x;
  int wave = tid >> 5, lane = tid & 31;
  int sel = lane >> 4, lr = lane & 15;
  int h = blockIdx.x, b = blockIdx.y;
  const unsigned short* qrow = qkv + ((size_t)b * O_QKV + h * 16 + lr) * NPIX;        // A: M=c rows
  const unsigned short* krow = qkv + ((size_t)b * O_QKV + 128 + h * 16 + lr) * NPIX;  // B col d = k row d
  v8f s;
#pragma unroll
  for (int i = 0; i < 8; ++i) s[i] = 0.f;
  int kbase = wave * 2048;                          // 8 waves split K=16384
  for (int it = 0; it < 64; ++it) {
    int kk = kbase + it * 32;
    V16 a, bm;
    a.u4[0]  = *(const uint4*)(qrow + kk + sel * 8);
    a.u4[1]  = *(const uint4*)(qrow + kk + sel * 8 + 16);
    bm.u4[0] = *(const uint4*)(krow + kk + sel * 8);
    bm.u4[1] = *(const uint4*)(krow + kk + sel * 8 + 16);
    s = __builtin_amdgcn_wmma_f32_16x16x32_bf16(false, a.v, false, bm.v,
                                                (short)0, s, false, false);
  }
#pragma unroll
  for (int r = 0; r < 8; ++r)
    spart[wave * 256 + (r + sel * 8) * 16 + lr] = s[r];
  __syncthreads();
  {                                                  // cross-wave reduce + fold l2norm & temperature
    float v = 0.f;
#pragma unroll
    for (int w2 = 0; w2 < 8; ++w2) v += spart[w2 * 256 + tid];
    int mrow = tid >> 4, dcol = tid & 15;
    float rq = 1.f / fmaxf(sqrtf(qknorm2[b * 256 + h * 16 + mrow]), 1e-12f);
    float rk = 1.f / fmaxf(sqrtf(qknorm2[b * 256 + 128 + h * 16 + dcol]), 1e-12f);
    sfin[tid] = v * temp_p[0] * rq * rk;
  }
  __syncthreads();
  if (tid < 16) {                                    // softmax over d (16)
    float mx = -1e30f;
    for (int d2 = 0; d2 < 16; ++d2) mx = fmaxf(mx, sfin[tid * 16 + d2]);
    float e[16], sum = 0.f;
    for (int d2 = 0; d2 < 16; ++d2) { e[d2] = __expf(sfin[tid * 16 + d2] - mx); sum += e[d2]; }
    float inv = 1.f / sum;
    for (int d2 = 0; d2 < 16; ++d2)
      P[((size_t)(b * NHEADS + h)) * 256 + tid * 16 + d2] = e[d2] * inv;
  }
}

// ---------------- out = P @ v (16x16 matvec per pixel, VALU; ~0.5 GF) ----------------
__global__ void attn_pv(const float* __restrict__ P,
                        const unsigned short* __restrict__ qkv,
                        unsigned short* __restrict__ out) {
  __shared__ float sp[256];
  int tid = threadIdx.x;
  int h = blockIdx.y, b = blockIdx.z;
  const float* psrc = &P[((size_t)(b * NHEADS + h)) * 256 + tid];
#ifdef USE_ASYNC_LDS
  __builtin_amdgcn_global_load_async_to_lds_b32(
      (AS_GLOBAL int*)psrc, (AS_LDS int*)&sp[tid], 0, 0);
  __builtin_amdgcn_s_wait_asynccnt(0);
#else
  sp[tid] = *psrc;
#endif
  __syncthreads();
  int n = blockIdx.x * 256 + tid;
  float acc[16];
#pragma unroll
  for (int c = 0; c < 16; ++c) acc[c] = 0.f;
  const unsigned short* vbase = qkv + ((size_t)b * O_QKV + 256 + h * 16) * NPIX + n;
#pragma unroll
  for (int d = 0; d < 16; ++d) {
    float vv = bf2f(vbase[(size_t)d * NPIX]);
#pragma unroll
    for (int c = 0; c < 16; ++c) acc[c] += sp[c * 16 + d] * vv;
  }
  unsigned short* obase = out + ((size_t)b * CCH + h * 16) * NPIX + n;
#pragma unroll
  for (int c = 0; c < 16; ++c) obase[(size_t)c * NPIX] = f2bf(acc[c]);
}

// ---------------- fused depthwise 3x3 (both halves) + exact GELU gate ----------------
__global__ void dw3_gate(const unsigned short* __restrict__ src,
                         const float* __restrict__ wdw,
                         unsigned short* __restrict__ g) {
  int tid = threadIdx.x;
  int c = blockIdx.y, b = blockIdx.z;
  int n = blockIdx.x * 256 + tid;
  size_t oidx = ((size_t)b * K_PWO + c) * NPIX + n;
  if (c >= MIDCH) { g[oidx] = 0; return; }          // zero K-padding rows for pwo GEMM
  int y = n >> 7, xx = n & 127;
  const unsigned short* r1 = src + ((size_t)b * O_PW12 + c) * NPIX;
  const unsigned short* r2 = src + ((size_t)b * O_PW12 + c + MIDCH) * NPIX;
  const float* w1 = wdw + c * 9;
  const float* w2 = wdw + (c + MIDCH) * 9;
  float a1 = 0.f, a2 = 0.f;
#pragma unroll
  for (int dy = -1; dy <= 1; ++dy) {
    int yy = y + dy;
    if (yy < 0 || yy > 127) continue;
#pragma unroll
    for (int dx = -1; dx <= 1; ++dx) {
      int xb = xx + dx;
      if (xb < 0 || xb > 127) continue;
      int w9 = (dy + 1) * 3 + (dx + 1);
      int pi = yy * 128 + xb;
      a1 += w1[w9] * bf2f(r1[pi]);
      a2 += w2[w9] * bf2f(r2[pi]);
    }
  }
  float ge = 0.5f * a1 * (1.f + erff(a1 * 0.70710678118f));   // exact GELU
  g[oidx] = f2bf(ge * a2);
}

// ---------------- host launch ----------------
static inline size_t alignup(size_t x) { return (x + 255) & ~(size_t)255; }

extern "C" void kernel_launch(void* const* d_in, const int* in_sizes, int n_in,
                              void* d_out, int out_size, void* d_ws, size_t ws_size,
                              hipStream_t stream) {
  (void)in_sizes; (void)n_in; (void)out_size;
  const float* x        = (const float*)d_in[0];
  const float* norm1_w  = (const float*)d_in[1];
  const float* norm1_b  = (const float*)d_in[2];
  const float* pwqkv_w  = (const float*)d_in[3];
  const float* dwqkv_w  = (const float*)d_in[4];
  const float* proj_w   = (const float*)d_in[5];
  const float* temp_p   = (const float*)d_in[6];
  const float* norm2_w  = (const float*)d_in[7];
  const float* norm2_b  = (const float*)d_in[8];
  const float* pw12_w   = (const float*)d_in[9];
  const float* dw12_w   = (const float*)d_in[10];
  const float* pwo_w    = (const float*)d_in[11];
  const float* attn_sc  = (const float*)d_in[12];
  const float* ff_sc    = (const float*)d_in[13];
  float* out = (float*)d_out;

  char* base = (char*)d_ws;
  size_t off = 0;
  auto carve = [&](size_t bytes) -> char* { char* p = base + off; off = alignup(off + bytes); return p; };

  // zero zone first (one memset covers all accumulators)
  float* stats1  = (float*)carve(BATCH * 2 * sizeof(float));
  float* stats2  = (float*)carve(BATCH * 2 * sizeof(float));
  float* qknorm2 = (float*)carve(BATCH * 256 * sizeof(float));
  size_t zero_bytes = off;

  unsigned short* wq_bf   = (unsigned short*)carve((size_t)O_QKV  * CCH   * 2);
  unsigned short* wpr_bf  = (unsigned short*)carve((size_t)CCH    * CCH   * 2);
  unsigned short* w12_bf  = (unsigned short*)carve((size_t)O_PW12 * CCH   * 2);
  unsigned short* wpo_bf  = (unsigned short*)carve((size_t)CCH    * K_PWO * 2);
  float*          Pbuf    = (float*)         carve((size_t)BATCH * NHEADS * 256 * 4);
  unsigned short* hnorm   = (unsigned short*)carve((size_t)BATCH * CCH   * NPIX * 2);
  unsigned short* attnout = (unsigned short*)carve((size_t)BATCH * CCH   * NPIX * 2);
  float*          x1      = (float*)         carve((size_t)BATCH * CCH   * NPIX * 4);
  size_t qkv_half = (size_t)BATCH * O_QKV * NPIX * 2;
  unsigned short* qkv_pw  = (unsigned short*)carve(2 * qkv_half);   // also reused as t_pw (688 rows fits in 2*384)
  unsigned short* qkv_dw  = qkv_pw + (size_t)BATCH * O_QKV * NPIX;
  unsigned short* t_pw    = qkv_pw;                                  // alias: qkv dead after proj
  unsigned short* gbuf    = (unsigned short*)carve((size_t)BATCH * K_PWO * NPIX * 2);
  (void)ws_size;

  (void)hipMemsetAsync(base, 0, zero_bytes, stream);

  // weight prep (fp32 -> bf16, padded)
  cvt_pad<<<dim3((O_QKV * CCH + 255) / 256), 256, 0, stream>>>(pwqkv_w, wq_bf,  O_QKV,  CCH,   O_QKV, CCH);
  cvt_pad<<<dim3((CCH * CCH + 255) / 256),   256, 0, stream>>>(proj_w,  wpr_bf, CCH,    CCH,   CCH,   CCH);
  cvt_pad<<<dim3((O_PW12 * CCH + 255) / 256),256, 0, stream>>>(pw12_w,  w12_bf, O_PW12, CCH,   680,   CCH);
  cvt_pad<<<dim3((CCH * K_PWO + 255) / 256), 256, 0, stream>>>(pwo_w,   wpo_bf, CCH,    K_PWO, CCH,   340);

  // ---- MDTA ----
  gn_stats<<<dim3(512, BATCH), 256, 0, stream>>>(x, stats1);
  gn_apply<<<dim3(16384), 256, 0, stream>>>(x, stats1, norm1_w, norm1_b, hnorm);
  gemm_wmma<<<dim3(NPIX / 64, O_QKV / 128, BATCH), 256, 0, stream>>>(
      wq_bf, hnorm, CCH, O_QKV, qkv_pw, nullptr, nullptr, nullptr, 0);
  dw3_qkv<<<dim3(NPIX / 256, O_QKV, BATCH), 256, 0, stream>>>(qkv_pw, dwqkv_w, qkv_dw, qknorm2);
  attn_qk<<<dim3(NHEADS, BATCH), 256, 0, stream>>>(qkv_dw, qknorm2, temp_p, Pbuf);
  attn_pv<<<dim3(NPIX / 256, NHEADS, BATCH), 256, 0, stream>>>(Pbuf, qkv_dw, attnout);
  gemm_wmma<<<dim3(NPIX / 64, 1, BATCH), 256, 0, stream>>>(
      wpr_bf, attnout, CCH, CCH, nullptr, x1, x, attn_sc, 1);

  // ---- GDFN ----
  gn_stats<<<dim3(512, BATCH), 256, 0, stream>>>(x1, stats2);
  gn_apply<<<dim3(16384), 256, 0, stream>>>(x1, stats2, norm2_w, norm2_b, hnorm);
  gemm_wmma<<<dim3(NPIX / 64, (O_PW12 + 127) / 128, BATCH), 256, 0, stream>>>(
      w12_bf, hnorm, CCH, O_PW12, t_pw, nullptr, nullptr, nullptr, 0);
  dw3_gate<<<dim3(NPIX / 256, K_PWO, BATCH), 256, 0, stream>>>(t_pw, dw12_w, gbuf);
  gemm_wmma<<<dim3(NPIX / 64, 1, BATCH), 256, 0, stream>>>(
      wpo_bf, gbuf, K_PWO, CCH, nullptr, out, x1, ff_sc, 1);
}
